// RNNDecoder_30923764531417
// MI455X (gfx1250) — compile-verified, matrix operand order checked by reading, not run on hardware
//
#include <hip/hip_runtime.h>

#define LATENT  128
#define DMODEL  512
#define ENC_IN  55
#define MTILE   16

typedef __attribute__((ext_vector_type(16))) __bf16 v16bf;
typedef __attribute__((ext_vector_type(8)))  float  v8f;

union BF16x16 { v16bf v; uint4 q[2]; };

__device__ __forceinline__ unsigned short f2bf(float f) {
    unsigned u = __float_as_uint(f);
    u += 0x7FFFu + ((u >> 16) & 1u);            // round-to-nearest-even
    return (unsigned short)(u >> 16);
}
__device__ __forceinline__ float sigmoid_(float x) { return 1.0f / (1.0f + __expf(-x)); }
__device__ __forceinline__ float tanh_(float x) {
    x = fminf(15.0f, fmaxf(-15.0f, x));
    float e = __expf(2.0f * x);
    return (e - 1.0f) / (e + 1.0f);
}

// ---------------- prep: f32 -> bf16 conversions (weights stay L2-resident) ----------------
__global__ void cvt_bf16_kernel(const float* __restrict__ src,
                                unsigned short* __restrict__ dst, int n) {
    int i = blockIdx.x * blockDim.x + threadIdx.x;
    if (i < n) dst[i] = f2bf(src[i]);
}
// out_w (55x512) -> padded (64x512) bf16, zero rows 55..63
__global__ void cvt_outw_kernel(const float* __restrict__ src,
                                unsigned short* __restrict__ dst) {
    int i = blockIdx.x * blockDim.x + threadIdx.x;   // 64*512 threads
    int r = i >> 9, c = i & 511;
    dst[i] = (r < ENC_IN) ? f2bf(src[r * DMODEL + c]) : (unsigned short)0;
}

// ---------------- main fused GRU kernel: one WG owns 16 batch rows ----------------
// Transposed WMMA roles: D[m=gate col, n=batch row] = sum_k W[m][k] * h[n][k]
//   A (SRC0) = weight rows (contiguous K), B (SRC1) = h state (batch on N axis).
// C/D layout then gives each lane a fixed batch row and 8 consecutive columns,
// so h lives in registers and the bf16 writeback is one packed ds_store_b128.
__global__ __launch_bounds__(512, 1) void gru_wmma_kernel(
    const unsigned short* __restrict__ z_bf,     // [B,128]   bf16
    const unsigned short* __restrict__ hproj_bf, // [512,128] bf16
    const float*          __restrict__ hproj_b,  // [512]
    const unsigned short* __restrict__ whh_bf,   // [1536,512] bf16
    const float*          __restrict__ b_ih,     // [1536]
    const float*          __restrict__ b_hh,     // [1536]
    const unsigned short* __restrict__ outw_bf,  // [64,512] bf16 (padded)
    const float*          __restrict__ out_b,    // [55]
    const int*            __restrict__ win_p,
    float*                __restrict__ out)      // [B, win, 55]
{
    __shared__ unsigned short hb[MTILE][528];    // bf16 state, [batch_row][k]

    const int win  = *win_p;
    const int wg   = blockIdx.x;                 // batch tile
    const int tid  = threadIdx.x;
    const int wave = tid >> 5;
    const int lane = tid & 31;
    const int nloc = lane & 15;                  // batch row (B/C n) or weight row (A m)
    const int hi   = lane >> 4;
    const int kb   = hi * 8;                     // per-lane K base (ISA operand layout)
    const int cb0  = wave * 2;                   // this wave's two column blocks

    float hreg[2][8];                            // exact f32 state, lane-owned

    // ---------- h0 = tanh(z @ Wp^T + b) : D[m=h0 col, n=batch row] ----------
    {
        const unsigned short* zrow = z_bf + (wg * MTILE + nloc) * LATENT;   // B: batch row
        #pragma unroll
        for (int blk = 0; blk < 2; ++blk) {
            const int cb = cb0 + blk;
            v8f acc = {};
            const unsigned short* arow = hproj_bf + (cb * 16 + nloc) * LATENT; // A: h0-col row
            #pragma unroll
            for (int k0 = 0; k0 < LATENT; k0 += 32) {
                BF16x16 a, b;
                a.q[0] = *(const uint4*)(arow + k0 + kb);
                a.q[1] = *(const uint4*)(arow + k0 + kb + 16);
                b.q[0] = *(const uint4*)(zrow + k0 + kb);
                b.q[1] = *(const uint4*)(zrow + k0 + kb + 16);
                acc = __builtin_amdgcn_wmma_f32_16x16x32_bf16(false, a.v, false, b.v,
                                                              (short)0, acc, false, false);
            }
            const int colbase = cb * 16 + hi * 8;
            uint4 pk; unsigned short* ps = (unsigned short*)&pk;
            #pragma unroll
            for (int e = 0; e < 8; ++e) {
                float h0 = tanh_(acc[e] + hproj_b[colbase + e]);
                hreg[blk][e] = h0;
                ps[e] = f2bf(h0);
            }
            *(uint4*)&hb[nloc][colbase] = pk;    // 8 consecutive bf16, 16B aligned
        }
    }

    // ---------- hoist all time-invariant per-lane data out of the loop ----------
    float bR[2][8], bZ[2][8], bIN[2][8], bHN[2][8];
    #pragma unroll
    for (int blk = 0; blk < 2; ++blk) {
        const int colbase = (cb0 + blk) * 16 + hi * 8;
        #pragma unroll
        for (int e = 0; e < 8; ++e) {
            const int c = colbase + e;
            bR[blk][e]  = b_ih[c] + b_hh[c];               // r gate: biases always summed
            bZ[blk][e]  = b_ih[512 + c] + b_hh[512 + c];   // z gate: biases always summed
            bIN[blk][e] = b_ih[1024 + c];
            bHN[blk][e] = b_hh[1024 + c];
        }
    }
    const unsigned short* wr0 = whh_bf + (        cb0 * 16 + nloc) * DMODEL;
    const unsigned short* wz0 = whh_bf + ( 512 +  cb0 * 16 + nloc) * DMODEL;
    const unsigned short* wn0 = whh_bf + (1024 +  cb0 * 16 + nloc) * DMODEL;
    const unsigned short* wr1 = wr0 + 16 * DMODEL;
    const unsigned short* wz1 = wz0 + 16 * DMODEL;
    const unsigned short* wn1 = wn0 + 16 * DMODEL;
    const unsigned short* hrow = &hb[nloc][0];   // shared B operand for all 6 gate WMMAs

    // output projection hoists (used by waves 0..3)
    const int ocb = wave * 16 + hi * 8;          // output column base for this lane
    float bo[8];
    #pragma unroll
    for (int e = 0; e < 8; ++e)
        bo[e] = (wave < 4 && (ocb + e) < ENC_IN) ? out_b[ocb + e] : 0.0f;
    const unsigned short* oarow = outw_bf + (wave * 16 + nloc) * DMODEL;  // A: out_w row
    float* orow = out + (long)(wg * MTILE + nloc) * win * ENC_IN;

    __syncthreads();

    // ---------- recurrence with fused per-step output projection ----------
    for (int t = 0; t < win; ++t) {
        v8f accR[2] = {{}, {}}, accZ[2] = {{}, {}}, accN[2] = {{}, {}};
        #pragma unroll 2
        for (int k0 = 0; k0 < DMODEL; k0 += 32) {
            BF16x16 hbt;                         // B operand: h state, reused 6x
            hbt.q[0] = *(const uint4*)(hrow + k0 + kb);
            hbt.q[1] = *(const uint4*)(hrow + k0 + kb + 16);
            BF16x16 a;
            a.q[0] = *(const uint4*)(wr0 + k0 + kb); a.q[1] = *(const uint4*)(wr0 + k0 + kb + 16);
            accR[0] = __builtin_amdgcn_wmma_f32_16x16x32_bf16(false, a.v, false, hbt.v, (short)0, accR[0], false, false);
            a.q[0] = *(const uint4*)(wz0 + k0 + kb); a.q[1] = *(const uint4*)(wz0 + k0 + kb + 16);
            accZ[0] = __builtin_amdgcn_wmma_f32_16x16x32_bf16(false, a.v, false, hbt.v, (short)0, accZ[0], false, false);
            a.q[0] = *(const uint4*)(wn0 + k0 + kb); a.q[1] = *(const uint4*)(wn0 + k0 + kb + 16);
            accN[0] = __builtin_amdgcn_wmma_f32_16x16x32_bf16(false, a.v, false, hbt.v, (short)0, accN[0], false, false);
            a.q[0] = *(const uint4*)(wr1 + k0 + kb); a.q[1] = *(const uint4*)(wr1 + k0 + kb + 16);
            accR[1] = __builtin_amdgcn_wmma_f32_16x16x32_bf16(false, a.v, false, hbt.v, (short)0, accR[1], false, false);
            a.q[0] = *(const uint4*)(wz1 + k0 + kb); a.q[1] = *(const uint4*)(wz1 + k0 + kb + 16);
            accZ[1] = __builtin_amdgcn_wmma_f32_16x16x32_bf16(false, a.v, false, hbt.v, (short)0, accZ[1], false, false);
            a.q[0] = *(const uint4*)(wn1 + k0 + kb); a.q[1] = *(const uint4*)(wn1 + k0 + kb + 16);
            accN[1] = __builtin_amdgcn_wmma_f32_16x16x32_bf16(false, a.v, false, hbt.v, (short)0, accN[1], false, false);
        }
        __syncthreads();                         // all gate reads of hb done

        #pragma unroll
        for (int blk = 0; blk < 2; ++blk) {
            const int colbase = (cb0 + blk) * 16 + hi * 8;
            uint4 pk; unsigned short* ps = (unsigned short*)&pk;
            #pragma unroll
            for (int e = 0; e < 8; ++e) {
                float r  = sigmoid_(bR[blk][e] + accR[blk][e]);
                float zg = sigmoid_(bZ[blk][e] + accZ[blk][e]);
                float n  = tanh_(bIN[blk][e] + r * (accN[blk][e] + bHN[blk][e]));
                float hn = (1.0f - zg) * n + zg * hreg[blk][e];
                hreg[blk][e] = hn;
                ps[e] = f2bf(hn);
            }
            *(uint4*)&hb[nloc][colbase] = pk;    // packed bf16 state writeback
        }
        __syncthreads();                         // new hb visible to all waves

        // fused y_t = h_t @ out_w^T + out_b : waves 0..3 own the 4 N-tiles
        if (wave < 4) {
            v8f acc = {};
            #pragma unroll 2
            for (int k0 = 0; k0 < DMODEL; k0 += 32) {
                BF16x16 a, b;
                a.q[0] = *(const uint4*)(oarow + k0 + kb);
                a.q[1] = *(const uint4*)(oarow + k0 + kb + 16);
                b.q[0] = *(const uint4*)(hrow + k0 + kb);
                b.q[1] = *(const uint4*)(hrow + k0 + kb + 16);
                acc = __builtin_amdgcn_wmma_f32_16x16x32_bf16(false, a.v, false, b.v,
                                                              (short)0, acc, false, false);
            }
            float* op = orow + (long)t * ENC_IN;
            #pragma unroll
            for (int e = 0; e < 8; ++e) {
                const int c = ocb + e;
                if (c < ENC_IN) op[c] = acc[e] + bo[e];
            }
        }
    }
}

extern "C" void kernel_launch(void* const* d_in, const int* in_sizes, int n_in,
                              void* d_out, int out_size, void* d_ws, size_t ws_size,
                              hipStream_t stream) {
    const float* z        = (const float*)d_in[0];
    const float* h_proj_w = (const float*)d_in[1];
    const float* h_proj_b = (const float*)d_in[2];
    /* d_in[3] = w_ih: matrix unused by the reference math (biases only)   */
    const float* w_hh     = (const float*)d_in[4];
    const float* b_ih     = (const float*)d_in[5];
    const float* b_hh     = (const float*)d_in[6];
    const float* out_w    = (const float*)d_in[7];
    const float* out_b    = (const float*)d_in[8];
    const int*   win_p    = (const int*)d_in[9];

    const int batch = in_sizes[0] / LATENT;              // 256

    unsigned short* ws_whh   = (unsigned short*)d_ws;                 // 1536*512
    unsigned short* ws_hproj = ws_whh   + 3 * DMODEL * DMODEL;        // 512*128
    unsigned short* ws_z     = ws_hproj + DMODEL * LATENT;            // B*128
    unsigned short* ws_outw  = ws_z     + batch * LATENT;             // 64*512

    cvt_bf16_kernel<<<(3 * DMODEL * DMODEL + 255) / 256, 256, 0, stream>>>(w_hh, ws_whh, 3 * DMODEL * DMODEL);
    cvt_bf16_kernel<<<(DMODEL * LATENT + 255) / 256, 256, 0, stream>>>(h_proj_w, ws_hproj, DMODEL * LATENT);
    cvt_bf16_kernel<<<(batch * LATENT + 255) / 256, 256, 0, stream>>>(z, ws_z, batch * LATENT);
    cvt_outw_kernel<<<(64 * DMODEL) / 256, 256, 0, stream>>>(out_w, ws_outw);

    gru_wmma_kernel<<<batch / MTILE, 512, 0, stream>>>(
        ws_z, ws_hproj, h_proj_b, ws_whh, b_ih, b_hh, ws_outw, out_b, win_p,
        (float*)d_out);
}
